// TransducerBeamSearcherMultitask_48567490183812
// MI455X (gfx1250) — matrix operand order repaired and checked by reading, not compiled, not run on hardware
//
#include <hip/hip_runtime.h>
#include <hip/hip_bf16.h>
#include <stdint.h>

typedef __attribute__((ext_vector_type(16))) _Float16 v16h;
typedef __attribute__((ext_vector_type(8)))  float    v8f;
typedef __attribute__((ext_vector_type(4))) unsigned  v4u;
typedef __attribute__((ext_vector_type(8))) int       v8i;
typedef __attribute__((ext_vector_type(4))) int       v4i;

#define GAS __attribute__((address_space(1)))

#define NWG    16
#define BLOCK  256
#define NWAVES ((NWG * BLOCK) / 32)   // 128 wave32s
#define NTHR   (NWG * BLOCK)          // 4096 threads

constexpr int Bn = 16, Tn = 150, Hn = 512, En = 512, Vn = 4096;
constexpr int KT      = Hn / 32;        // 16 K-tiles for every matmul (K=512)
constexpr int TILE_H  = KT * 32 * 16;   // halves per packed 512x16 B column-tile = 8192

// ---------------- workspace layout (bytes) ----------------
constexpr size_t OFF_CTR  = 0;                       // grid-barrier slots
constexpr size_t CTR_BYTES= 4096;                    // 1024 u32 slots (need 752)
constexpr size_t OFF_H    = 4096;                    // h[2][16][512] f32
constexpr size_t OFF_TOK  = OFF_H   + 2*16*512*4;    // tok[2][16] i32
constexpr size_t OFF_MSK  = OFF_TOK + 128;           // mask[2][16] i32
constexpr size_t OFF_LP   = OFF_MSK + 128;           // lp[2][16] f32
constexpr size_t OFF_XA   = 70144;                   // A-pack: (tn_t+h_a+h_p) 16x512 f16
constexpr size_t SZ_AP    = 16384;                   // 16*512 halves * 2B
constexpr size_t OFF_JA   = OFF_XA  + SZ_AP;         // A-pack: joint
constexpr size_t OFF_EA0  = OFF_JA  + SZ_AP;         // A-pack: emb[tok] asr   (EA0,EA1,HA0,HA1 contiguous)
constexpr size_t OFF_EA1  = OFF_EA0 + SZ_AP;         // A-pack: emb[tok] pos
constexpr size_t OFF_HA0  = OFF_EA1 + SZ_AP;         // A-pack: h asr
constexpr size_t OFF_HA1  = OFF_HA0 + SZ_AP;         // A-pack: h pos
constexpr size_t OFF_LOG  = OFF_HA1 + SZ_AP;         // logits[2][16][4096] f32
constexpr size_t OFF_WJ   = OFF_LOG + 2*16*4096*4;   // packed wj   512x512  f16
constexpr size_t OFF_WC0  = OFF_WJ  + 512*512*2;     // packed wc_asr 512x4096
constexpr size_t OFF_WC1  = OFF_WC0 + 512*4096*2;    // packed wc_pos
constexpr size_t OFF_WI0  = OFF_WC1 + 512*4096*2;    // packed wi_asr^T 512x1536
constexpr size_t OFF_WI1  = OFF_WI0 + 512*1536*2;
constexpr size_t OFF_WH0  = OFF_WI1 + 512*1536*2;    // packed wh_asr^T 512x1536
constexpr size_t OFF_WH1  = OFF_WH0 + 512*1536*2;

// ---------------- layout helpers (CDNA5 WMMA VGPR layouts, ISA 7.12.2) ----------------
// 16-bit A matrix 16x32: lane l, elem e -> m = l%16 ; k = 8*(l/16) + (e<8 ? e : 8+e)
__device__ __host__ inline int a_elem_index(int m, int k) {
  int kt = k >> 5, kk = k & 31;
  int lane = m + (((kk >> 3) & 1) << 4);
  int e = (kk & 7) + ((kk & 16) ? 8 : 0);
  return (kt * 32 + lane) * 16 + e;
}

// ---------------- weight pre-pack: f32 -> f16 B-operand tiles ----------------
// dst layout: [N/16][K/32][lane=32][e=16] halves; B[k,n]: lane = n%16 + 16*((k%32/8)&1), e = k%8 + 8*(k%32>=16)
__global__ void pack_b_kernel(const float* __restrict__ src, _Float16* __restrict__ dst,
                              int K, int N, int srcTransposed) {
  int total = (N >> 4) * (K >> 5) * 512;
  int ktiles = K >> 5;
  const GAS float* s = (const GAS float*)src;
  GAS _Float16* d = (GAS _Float16*)dst;
  for (int i = blockIdx.x * blockDim.x + threadIdx.x; i < total; i += gridDim.x * blockDim.x) {
    int e    = i & 15;
    int lane = (i >> 4) & 31;
    int kt   = (i >> 9) % ktiles;
    int nt   = (i >> 9) / ktiles;
    int n = nt * 16 + (lane & 15);
    int k = kt * 32 + ((lane >> 4) << 3) + (e < 8 ? e : 8 + e);
    float v = srcTransposed ? s[(size_t)n * K + k] : s[(size_t)k * N + n];
    d[i] = (_Float16)v;
  }
}

// ---------------- Tensor Data Mover: 1-D global -> LDS copy (ISA ch.8 D# layout) ----------------
// data_size=8B elements; n8 = number of 8-byte elements (<= 65535)
__device__ inline void tdm_load_1d(const void* gsrc, unsigned lds_byte_off, unsigned n8) {
  unsigned long long ga = (unsigned long long)(uintptr_t)gsrc;
  v4u g0 = { 1u,                                    // count=1 valid, user mode
             lds_byte_off,                          // D#.lds_addr
             (unsigned)(ga & 0xffffffffull),        // global_addr[31:0]
             (unsigned)((ga >> 32) & 0x1ffffffull) | (2u << 30) };  // [56:32] | type=2
  v8i g1 = { (int)(3u << 16),                       // wg_mask=0, data_size=3 (8B)
             (int)(n8 << 16),                       // tensor_dim0[15:0] (bits 63:48)
             (int)((n8 >> 16) | (1u << 16)),        // tensor_dim0[31:16]; tensor_dim1=1
             (int)(n8 << 16),                       // tile_dim0 (bits 127:112)
             1,                                     // tile_dim1=1
             (int)n8,                               // tensor_dim0_stride[31:0]
             0, 0 };
  v4i gz = { 0, 0, 0, 0 };
#if defined(__clang_major__) && (__clang_major__ >= 23)
  v8i gz8 = { 0, 0, 0, 0, 0, 0, 0, 0 };
  __builtin_amdgcn_tensor_load_to_lds(g0, g1, gz, gz, gz8, 0);
#else
  __builtin_amdgcn_tensor_load_to_lds(g0, g1, gz, gz, 0);
#endif
}

// ---------------- device helpers ----------------
__device__ inline void grid_sync(unsigned* ctr, int slot, int tid) {
  __threadfence();
  __syncthreads();
  if (tid == 0) {
    __hip_atomic_fetch_add(&ctr[slot], 1u, __ATOMIC_RELEASE, __HIP_MEMORY_SCOPE_AGENT);
    while (__hip_atomic_load(&ctr[slot], __ATOMIC_ACQUIRE, __HIP_MEMORY_SCOPE_AGENT) < (unsigned)NWG) {
      __builtin_amdgcn_s_sleep(2);
    }
  }
  __syncthreads();
}

__device__ inline v8f bias_init(const float* bias, int n0, int lane) {
  const GAS float* b = (const GAS float*)bias;
  float bv = b[n0 + (lane & 15)];
  v8f a;
#pragma unroll
  for (int i = 0; i < 8; ++i) a[i] = bv;
  return a;
}

// D(16x16,f32) += A(16x512,f16 from LDS) * B(512x16,f16 packed global tile)
// one-stage software pipeline: preload tile kt+1 while WMMA kt executes
__device__ inline v8f mm_tile(const _Float16* As /* LDS staged */,
                              const _Float16* Bp /* global */,
                              int lane, v8f acc) {
  const v16h* A  = reinterpret_cast<const v16h*>(As);
  const GAS v16h* Bv = (const GAS v16h*)(const v16h*)Bp;
  __builtin_prefetch(Bp + TILE_H, 0, 1);   // next weight tile
  v16h a = A[lane];
  v16h b = Bv[lane];
#pragma unroll
  for (int kt = 0; kt < KT; ++kt) {
    v16h an = a, bn = b;
    if (kt < KT - 1) {
      an = A[(kt + 1) * 32 + lane];
      bn = Bv[(kt + 1) * 32 + lane];
    }
    acc = __builtin_amdgcn_wmma_f32_16x16x32_f16(false, a, false, b, (short)0, acc, false, false);
    a = an; b = bn;
  }
  return acc;
}

__device__ inline float sigmoidf_(float x) { return 1.f / (1.f + __expf(-x)); }

// GRU phase: 64 jobs = (task, col-tile ct of H). SA = LDS copy of [EA0|EA1|HA0|HA1] (8192 halves each).
__device__ void gru_phase(int wave, int lane, const _Float16* SA,
                          const _Float16* WI0, const _Float16* WI1,
                          const _Float16* WH0, const _Float16* WH1,
                          const float* bi0, const float* bi1,
                          const float* bh0, const float* bh1,
                          float* hbuf, const int* mskb) {
  for (int job = wave; job < 64; job += NWAVES) {
    int task = job >> 5, ct = job & 31;
    const _Float16* EA = SA + (size_t)task * 8192;
    const _Float16* HA = SA + 16384 + (size_t)task * 8192;
    const _Float16* WI = task ? WI1 : WI0;
    const _Float16* WH = task ? WH1 : WH0;
    const float* bi = task ? bi1 : bi0;
    const float* bh = task ? bh1 : bh0;

    v8f ir = bias_init(bi,        ct * 16, lane);
    ir = mm_tile(EA, WI + (size_t)(ct)       * TILE_H, lane, ir);
    v8f iz = bias_init(bi,  512 + ct * 16, lane);
    iz = mm_tile(EA, WI + (size_t)(32 + ct)  * TILE_H, lane, iz);
    v8f in_ = bias_init(bi, 1024 + ct * 16, lane);
    in_ = mm_tile(EA, WI + (size_t)(64 + ct) * TILE_H, lane, in_);
    v8f hr = bias_init(bh,        ct * 16, lane);
    hr = mm_tile(HA, WH + (size_t)(ct)       * TILE_H, lane, hr);
    v8f hz = bias_init(bh,  512 + ct * 16, lane);
    hz = mm_tile(HA, WH + (size_t)(32 + ct)  * TILE_H, lane, hz);
    v8f hn = bias_init(bh, 1024 + ct * 16, lane);
    hn = mm_tile(HA, WH + (size_t)(64 + ct) * TILE_H, lane, hn);

#pragma unroll
    for (int r8 = 0; r8 < 8; ++r8) {
      int m = r8 + ((lane >> 4) << 3);
      int n = ct * 16 + (lane & 15);
      float rg = sigmoidf_(ir[r8] + hr[r8]);
      float zg = sigmoidf_(iz[r8] + hz[r8]);
      float ng = tanhf(in_[r8] + rg * hn[r8]);
      GAS float* hp = (GAS float*)(hbuf + ((size_t)(task * 16 + m)) * Hn + n);
      float hold = *hp;
      float hnew = (1.f - zg) * ng + zg * hold;
      *hp = mskb[task * 16 + m] ? hnew : hold;
    }
  }
}

// ---------------- persistent decode kernel: 16 WGs, whole T loop inside ----------------
__global__ __launch_bounds__(BLOCK, 1)
void decode_kernel(const float* __restrict__ tn, const float* __restrict__ bj,
                   const float* __restrict__ emb_a, const float* __restrict__ bi_a,
                   const float* __restrict__ bh_a,  const float* __restrict__ bc_a,
                   const float* __restrict__ emb_p, const float* __restrict__ bi_p,
                   const float* __restrict__ bh_p,  const float* __restrict__ bc_p,
                   char* __restrict__ ws, float* __restrict__ out) {
  // A-operand staging buffer, filled by the Tensor Data Mover. Single LDS object -> offset 0.
  __shared__ __align__(128) _Float16 smemA[4 * 8192];   // 64 KB

  unsigned* ctr = (unsigned*)(ws + OFF_CTR);
  float* hbuf = (float*)(ws + OFF_H);
  int*   tokb = (int*)(ws + OFF_TOK);
  int*   mskb = (int*)(ws + OFF_MSK);
  float* lpb  = (float*)(ws + OFF_LP);
  GAS _Float16* XA  = (GAS _Float16*)(ws + OFF_XA);
  GAS _Float16* JA  = (GAS _Float16*)(ws + OFF_JA);
  GAS _Float16* EA0 = (GAS _Float16*)(ws + OFF_EA0);
  GAS _Float16* EA1 = (GAS _Float16*)(ws + OFF_EA1);
  GAS _Float16* HA0 = (GAS _Float16*)(ws + OFF_HA0);
  GAS _Float16* HA1 = (GAS _Float16*)(ws + OFF_HA1);
  float* logits = (float*)(ws + OFF_LOG);
  const _Float16* WJ  = (const _Float16*)(ws + OFF_WJ);
  const _Float16* WC0 = (const _Float16*)(ws + OFF_WC0);
  const _Float16* WC1 = (const _Float16*)(ws + OFF_WC1);
  const _Float16* WI0 = (const _Float16*)(ws + OFF_WI0);
  const _Float16* WI1 = (const _Float16*)(ws + OFF_WI1);
  const _Float16* WH0 = (const _Float16*)(ws + OFF_WH0);
  const _Float16* WH1 = (const _Float16*)(ws + OFF_WH1);

  const int tid  = threadIdx.x;
  const int lane = tid & 31;
  const int wave = blockIdx.x * (BLOCK / 32) + (tid >> 5);
  const int gtid = blockIdx.x * BLOCK + tid;
  int bar = 0;

  // Stage a contiguous ws region into LDS via TDM; wave0 issues & waits TENSORcnt, then WG barrier.
  // IMPORTANT: the TDM intrinsic's LDS destination is descriptor *data*, invisible to alias
  // analysis, and smemA's address otherwise never escapes — so the compiler would fold all
  // reads of the never-"stored" LDS to undef. Escaping smemA's address into a may-write asm
  // (pointer operand + memory clobber) forces real ds_load traffic after each stage.
  auto stage = [&](const void* src, unsigned bytes) {
    if ((tid >> 5) == 0) {
      tdm_load_1d(src, 0u, bytes >> 3);
      __builtin_amdgcn_s_wait_tensorcnt(0);
    }
    __syncthreads();
    void* p = (void*)smemA;
    asm volatile("" : "+v"(p) : : "memory");
    (void)p;
  };

  // ---- init: h=0, tok=BLANK, lp=0, mask=1; pack EA from emb[BLANK], HA=0 ----
  for (int i = gtid; i < 2 * Bn * Hn; i += NTHR) ((GAS float*)hbuf)[i] = 0.f;
  for (int i = gtid; i < 32; i += NTHR) { tokb[i] = 0; mskb[i] = 1; lpb[i] = 0.f; }
  for (int i = gtid; i < 2 * Bn * En; i += NTHR) {
    int task = i >> 13, rem = i & 8191, m = rem >> 9, k = rem & 511;
    const GAS float* er = (const GAS float*)(task ? emb_p : emb_a);  // token 0 (BLANK) row
    int ai = a_elem_index(m, k);
    (task ? EA1 : EA0)[ai] = (_Float16)er[k];
    (task ? HA1 : HA0)[ai] = (_Float16)0.f;
  }
  grid_sync(ctr, bar++, tid);
  stage((const _Float16*)EA0, 4 * SZ_AP);              // EA0|EA1|HA0|HA1 contiguous, 64 KB
  gru_phase(wave, lane, smemA, WI0, WI1, WH0, WH1,
            bi_a, bi_p, bh_a, bh_p, hbuf, mskb);       // h_a0, h_p0
  grid_sync(ctr, bar++, tid);

  // ---- main sequential decode loop ----
  for (int t = 0; t < Tn; ++t) {
    // P1: x = tn_t + h_a + h_p, pack into A-operand layout (f16)
    for (int i = gtid; i < Bn * Hn; i += NTHR) {
      int m = i >> 9, k = i & 511;
      float x = ((const GAS float*)tn)[((size_t)m * Tn + t) * Hn + k]
              + ((const GAS float*)hbuf)[(size_t)m * Hn + k]
              + ((const GAS float*)hbuf)[(size_t)(16 + m) * Hn + k];
      XA[a_elem_index(m, k)] = (_Float16)x;
    }
    grid_sync(ctr, bar++, tid);

    // P2: joint = leaky_relu(x @ wj + bj); repack as A operand
    stage((const _Float16*)XA, SZ_AP);
    for (int job = wave; job < 32; job += NWAVES) {
      v8f acc = bias_init(bj, job * 16, lane);
      acc = mm_tile(smemA, WJ + (size_t)job * TILE_H, lane, acc);
#pragma unroll
      for (int r8 = 0; r8 < 8; ++r8) {
        int m = r8 + ((lane >> 4) << 3);
        int n = job * 16 + (lane & 15);
        float v = acc[r8];
        v = v > 0.f ? v : 0.01f * v;
        JA[a_elem_index(m, n)] = (_Float16)v;
      }
    }
    grid_sync(ctr, bar++, tid);

    // P3: logits = joint @ wc + bc for both tasks (512 tile jobs)
    stage((const _Float16*)JA, SZ_AP);
    for (int job = wave; job < 512; job += NWAVES) {
      int task = job >> 8, nt = job & 255;
      const _Float16* WC = task ? WC1 : WC0;
      const float* bc = task ? bc_p : bc_a;
      v8f acc = bias_init(bc, nt * 16, lane);
      acc = mm_tile(smemA, WC + (size_t)nt * TILE_H, lane, acc);
#pragma unroll
      for (int r8 = 0; r8 < 8; ++r8) {
        int m = r8 + ((lane >> 4) << 3);
        ((GAS float*)logits)[((size_t)task * Bn + m) * Vn + nt * 16 + (lane & 15)] = acc[r8];
      }
    }
    grid_sync(ctr, bar++, tid);

    // P4: per-(task,row) argmax + logsumexp, token select, emb gather, A-repacks
    if (wave < 32) {
      int task = wave >> 4, m = wave & 15;
      const GAS float* lg = (const GAS float*)(logits + ((size_t)task * Bn + m) * Vn);
      float best = -3.0e38f; int bidx = 0;
      for (int v = lane; v < Vn; v += 32) {
        float x = lg[v];
        if (x > best) { best = x; bidx = v; }
      }
      for (int off = 16; off > 0; off >>= 1) {
        float ob = __shfl_xor(best, off);
        int   oi = __shfl_xor(bidx, off);
        if (ob > best || (ob == best && oi < bidx)) { best = ob; bidx = oi; }
      }
      float sum = 0.f;
      for (int v = lane; v < Vn; v += 32) sum += __expf(lg[v] - best);
      for (int off = 16; off > 0; off >>= 1) sum += __shfl_xor(sum, off);
      float val = -__logf(sum);                 // max log-softmax value
      int msk = (bidx != 0);
      int tok_n = msk ? bidx : tokb[task * 16 + m];
      if (lane == 0) {
        tokb[task * 16 + m] = tok_n;
        mskb[task * 16 + m] = msk;
        lpb[task * 16 + m] += msk ? val : 0.f;
        ((GAS float*)out)[(size_t)task * Bn * Tn + (size_t)m * Tn + t] = msk ? (float)bidx : 0.f;
      }
      const GAS float* embrow = (const GAS float*)((task ? emb_p : emb_a) + (size_t)tok_n * En);
      const GAS float* hrow = (const GAS float*)(hbuf + ((size_t)task * Bn + m) * Hn);
      GAS _Float16* EA = task ? EA1 : EA0;
      GAS _Float16* HA = task ? HA1 : HA0;
      for (int k = lane; k < En; k += 32) {
        int ai = a_elem_index(m, k);
        EA[ai] = (_Float16)embrow[k];
        HA[ai] = (_Float16)hrow[k];
      }
    }
    grid_sync(ctr, bar++, tid);

    // P5: both GRU cells, masked h update
    stage((const _Float16*)EA0, 4 * SZ_AP);      // 64 KB: EA0|EA1|HA0|HA1
    gru_phase(wave, lane, smemA, WI0, WI1, WH0, WH1,
              bi_a, bi_p, bh_a, bh_p, hbuf, mskb);
    grid_sync(ctr, bar++, tid);
  }

  // scores [2,B]
  if (wave == 0 && lane < 32) {
    int task = lane >> 4, m = lane & 15;
    ((GAS float*)out)[(size_t)2 * Bn * Tn + task * Bn + m] = lpb[task * 16 + m];
  }
}

// ---------------- host entry ----------------
extern "C" void kernel_launch(void* const* d_in, const int* in_sizes, int n_in,
                              void* d_out, int out_size, void* d_ws, size_t ws_size,
                              hipStream_t stream) {
  (void)in_sizes; (void)n_in; (void)out_size; (void)ws_size;
  const float* tn    = (const float*)d_in[0];
  const float* wj    = (const float*)d_in[1];
  const float* bj    = (const float*)d_in[2];
  const float* emb_a = (const float*)d_in[3];
  const float* wi_a  = (const float*)d_in[4];
  const float* wh_a  = (const float*)d_in[5];
  const float* bi_a  = (const float*)d_in[6];
  const float* bh_a  = (const float*)d_in[7];
  const float* wc_a  = (const float*)d_in[8];
  const float* bc_a  = (const float*)d_in[9];
  const float* emb_p = (const float*)d_in[10];
  const float* wi_p  = (const float*)d_in[11];
  const float* wh_p  = (const float*)d_in[12];
  const float* bi_p  = (const float*)d_in[13];
  const float* bh_p  = (const float*)d_in[14];
  const float* wc_p  = (const float*)d_in[15];
  const float* bc_p  = (const float*)d_in[16];
  char* ws = (char*)d_ws;
  float* out = (float*)d_out;

  hipMemsetAsync(ws + OFF_CTR, 0, CTR_BYTES, stream);

  // one-time weight packing (f32 -> f16 WMMA B-tiles)
  pack_b_kernel<<<512,  256, 0, stream>>>(wj,   (_Float16*)(ws + OFF_WJ),  512, 512,  0);
  pack_b_kernel<<<2048, 256, 0, stream>>>(wc_a, (_Float16*)(ws + OFF_WC0), 512, 4096, 0);
  pack_b_kernel<<<2048, 256, 0, stream>>>(wc_p, (_Float16*)(ws + OFF_WC1), 512, 4096, 0);
  pack_b_kernel<<<1024, 256, 0, stream>>>(wi_a, (_Float16*)(ws + OFF_WI0), 512, 1536, 1);
  pack_b_kernel<<<1024, 256, 0, stream>>>(wi_p, (_Float16*)(ws + OFF_WI1), 512, 1536, 1);
  pack_b_kernel<<<1024, 256, 0, stream>>>(wh_a, (_Float16*)(ws + OFF_WH0), 512, 1536, 1);
  pack_b_kernel<<<1024, 256, 0, stream>>>(wh_p, (_Float16*)(ws + OFF_WH1), 512, 1536, 1);

  decode_kernel<<<NWG, BLOCK, 0, stream>>>(tn, bj,
                                           emb_a, bi_a, bh_a, bc_a,
                                           emb_p, bi_p, bh_p, bc_p,
                                           ws, out);
}
